// Attention_3590592660019
// MI455X (gfx1250) — matrix operand order repaired and checked by reading, not compiled, not run on hardware
//
#include <hip/hip_runtime.h>
#include <math.h>

// ---------------- problem constants ----------------
#define BB   2
#define SS   2048
#define DIMD 2048
#define HH   16
#define KVH  8
#define HD   128
#define MTOT (BB * SS)          // 4096
#define NQ   (HH * HD)          // 2048
#define NKV  (KVH * HD)         // 1024
#define NQKV (NQ + 2 * NKV)     // 4096 fused QKV output width
#define HALF (HD / 2)           // 64

typedef __attribute__((ext_vector_type(16))) __bf16        v16bf;
typedef __attribute__((ext_vector_type(8)))  float         v8f;
typedef __attribute__((ext_vector_type(4)))  unsigned int  v4u;
typedef __attribute__((ext_vector_type(4)))  int           v4i;

union Frag {
    v16bf bf;
    v4u   q[2];
};

__device__ __forceinline__ unsigned short f2b(float f) {
    unsigned u = __float_as_uint(f);
    u += 0x7fffu + ((u >> 16) & 1u);   // round-to-nearest-even
    return (unsigned short)(u >> 16);
}
__device__ __forceinline__ float b2f(unsigned short h) {
    return __uint_as_float(((unsigned)h) << 16);
}

// ---- gfx1250 async global->LDS copy (guarded; safe fallback) ----
#if defined(__has_builtin)
#if __has_builtin(__builtin_amdgcn_global_load_async_to_lds_b128)
#define HAVE_ASYNC_LDS 1
#endif
#endif
#ifndef HAVE_ASYNC_LDS
#define HAVE_ASYNC_LDS 0
#endif

typedef __attribute__((address_space(1))) v4i* gvp4;
typedef __attribute__((address_space(3))) v4i* lvp4;

__device__ __forceinline__ void lds_copy16(const unsigned short* g, unsigned short* l) {
#if HAVE_ASYNC_LDS
    __builtin_amdgcn_global_load_async_to_lds_b128((gvp4)g, (lvp4)l, 0, 0);
#else
    *(v4u*)l = *(const v4u*)g;
#endif
}
__device__ __forceinline__ void wait_lds_copies() {
#if HAVE_ASYNC_LDS
#if __has_builtin(__builtin_amdgcn_s_wait_asynccnt)
    __builtin_amdgcn_s_wait_asynccnt(0);
#else
    asm volatile("s_wait_asynccnt 0" ::: "memory");
#endif
#endif
}

// ---------------- fp32 -> bf16 (row-major copy) ----------------
__global__ void conv_kernel(const float* __restrict__ src,
                            unsigned short* __restrict__ dst, long n) {
    long i = (long)blockIdx.x * blockDim.x + threadIdx.x;
    if (i < n) dst[i] = f2b(src[i]);
}

// ---------------- fp32 [K][N] -> bf16 transposed [N][K] ----------------
__global__ void convT_kernel(const float* __restrict__ src,
                             unsigned short* __restrict__ dst, int K, int N) {
    long i = (long)blockIdx.x * blockDim.x + threadIdx.x;
    long tot = (long)K * N;
    if (i >= tot) return;
    int k = (int)(i % K);
    int n = (int)(i / K);
    dst[(size_t)n * K + k] = f2b(src[(size_t)k * N + n]);
}

// ---------------- bf16 GEMM: C[M][N] = A[M][K] * B^T[N][K] ----------------
// block = 256 threads = 8 waves; block tile 128(M) x 128(N); wave tile 32x64.
template <int OUT_BF16>
__global__ __launch_bounds__(256) void gemm_bf16(const unsigned short* __restrict__ A,
                                                 const unsigned short* __restrict__ BT,
                                                 void* __restrict__ Cout,
                                                 int M, int N, int K) {
    const int lane = threadIdx.x & 31;
    const int wave = threadIdx.x >> 5;
    const int wm   = wave & 3;          // 0..3 -> 32-row strip
    const int wn   = wave >> 2;         // 0..1 -> 64-col strip
    const int l15  = lane & 15;
    const int lh   = lane >> 4;
    const int rowBase = blockIdx.y * 128 + wm * 32;
    const int colBase = blockIdx.x * 128 + wn * 64;

    v8f c[2][4];
#pragma unroll
    for (int mi = 0; mi < 2; ++mi)
#pragma unroll
        for (int ni = 0; ni < 4; ++ni) c[mi][ni] = {};

    for (int koff = 0; koff < K; koff += 32) {
        Frag a[2], b[4];
#pragma unroll
        for (int mi = 0; mi < 2; ++mi) {
            const unsigned short* p =
                A + (size_t)(rowBase + mi * 16 + l15) * K + koff + lh * 8;
            a[mi].q[0] = *(const v4u*)p;          // K = kbase .. kbase+7
            a[mi].q[1] = *(const v4u*)(p + 16);   // K = 16+kbase .. 23+kbase
        }
#pragma unroll
        for (int ni = 0; ni < 4; ++ni) {
            const unsigned short* p =
                BT + (size_t)(colBase + ni * 16 + l15) * K + koff + lh * 8;
            b[ni].q[0] = *(const v4u*)p;          // K = lh*8 .. +7
            b[ni].q[1] = *(const v4u*)(p + 16);   // K = 16+lh*8 .. +7
        }
        if (koff + 32 < K)   // pull next A strip toward the caches (global_prefetch_b8)
            __builtin_prefetch(A + (size_t)(rowBase + l15) * K + koff + 32, 0, 1);
#pragma unroll
        for (int mi = 0; mi < 2; ++mi)
#pragma unroll
            for (int ni = 0; ni < 4; ++ni)
                c[mi][ni] = __builtin_amdgcn_wmma_f32_16x16x32_bf16(
                    false, a[mi].bf, false, b[ni].bf, (short)0, c[mi][ni],
                    false, false);
    }

#pragma unroll
    for (int mi = 0; mi < 2; ++mi)
#pragma unroll
        for (int ni = 0; ni < 4; ++ni)
#pragma unroll
            for (int r = 0; r < 8; ++r) {
                int row = rowBase + mi * 16 + r + (lh << 3);
                int col = colBase + ni * 16 + l15;
                float v = c[mi][ni][r];
                if (OUT_BF16)
                    ((unsigned short*)Cout)[(size_t)row * N + col] = f2b(v);
                else
                    ((float*)Cout)[(size_t)row * N + col] = v;
            }
}

// ---------------- RoPE + rearrange: fused-QKV rows -> [B][nh][S][HD] ----------------
__global__ void rope_kernel(const unsigned short* __restrict__ src, int srcStride, int colOff,
                            const float* __restrict__ cosf,
                            const float* __restrict__ sinf,
                            unsigned short* __restrict__ dst, int nh) {
    long i = (long)blockIdx.x * blockDim.x + threadIdx.x;
    long tot = (long)BB * SS * nh * HALF;
    if (i >= tot) return;
    int  dp = (int)(i % HALF);
    long r  = i / HALF;
    int  h  = (int)(r % nh);  r /= nh;
    int  s  = (int)(r % SS);
    int  b  = (int)(r / SS);
    float c  = cosf[s * HALF + dp];
    float sn = sinf[s * HALF + dp];
    size_t so = (size_t)(b * SS + s) * srcStride + colOff + h * HD + 2 * dp;
    float x0 = b2f(src[so]);
    float x1 = b2f(src[so + 1]);
    size_t doff = (((size_t)b * nh + h) * SS + s) * (size_t)HD + 2 * dp;
    dst[doff]     = f2b(x0 * c - x1 * sn);
    dst[doff + 1] = f2b(x0 * sn + x1 * c);
}

// ---------------- V rearrange: fused-QKV rows -> [B][KV][HD][S] (transposed) ----------------
__global__ void vtrans_kernel(const unsigned short* __restrict__ src, int srcStride, int colOff,
                              unsigned short* __restrict__ dst) {
    long i = (long)blockIdx.x * blockDim.x + threadIdx.x;
    long tot = (long)BB * SS * KVH * HD;
    if (i >= tot) return;
    int  d  = (int)(i % HD);
    long r  = i / HD;
    int  kv = (int)(r % KVH); r /= KVH;
    int  s  = (int)(r % SS);
    int  b  = (int)(r / SS);
    dst[(((size_t)b * KVH + kv) * HD + d) * SS + s] =
        src[(size_t)(b * SS + s) * srcStride + colOff + kv * HD + d];
}

// ---------------- causal flash attention ----------------
// block = 128 threads = 4 waves; block covers 64 queries (16 per wave).
// K/V chunks (32 keys) are staged cooperatively in LDS and shared by all waves.
// Qa: [B][H][S][HD], Ka: [B][KV][S][HD], Vt: [B][KV][HD][S], att: [M][DIM] bf16.
__global__ __launch_bounds__(128) void attn_kernel(const unsigned short* __restrict__ Qa,
                                                   const unsigned short* __restrict__ Ka,
                                                   const unsigned short* __restrict__ Vt,
                                                   unsigned short* __restrict__ att) {
    __shared__ __attribute__((aligned(32))) unsigned short Kl[32 * HD];   // 8 KB
    __shared__ __attribute__((aligned(32))) unsigned short Vl[HD * 32];   // 8 KB
    __shared__ __attribute__((aligned(32))) unsigned short pbuf[4][16 * 32]; // 4 KB

    const int tid  = threadIdx.x;
    const int lane = tid & 31;
    const int w    = tid >> 5;
    const int l15  = lane & 15;
    const int lh   = lane >> 4;
    const int bh   = blockIdx.y;
    const int b    = bh / HH;
    const int h    = bh % HH;
    const int kv   = h >> 1;                 // N_REP = 2
    const int q0blk = blockIdx.x * 64;
    const int q0    = q0blk + w * 16;

    const unsigned short* qbase = Qa + ((size_t)(b * HH + h) * SS) * HD;
    const unsigned short* kbase = Ka + ((size_t)(b * KVH + kv) * SS) * HD;
    const unsigned short* vbase = Vt + ((size_t)(b * KVH + kv) * HD) * SS;
    const float scale = 0.08838834764831845f;  // 1/sqrt(128)

    // Q fragments: 16 x 128 bf16, A-operand layout
    Frag aq[4];
#pragma unroll
    for (int t = 0; t < 4; ++t) {
        const unsigned short* p = qbase + (size_t)(q0 + l15) * HD + t * 32 + lh * 8;
        aq[t].q[0] = *(const v4u*)p;
        aq[t].q[1] = *(const v4u*)(p + 16);
    }

    v8f   o[8];
    float l[8], mrow[8], alpha[8];
#pragma unroll
    for (int t = 0; t < 8; ++t) o[t] = {};
#pragma unroll
    for (int r = 0; r < 8; ++r) { l[r] = 0.f; mrow[r] = -__builtin_inff(); }

    const int nchunks = (q0blk + 64) / 32;   // uniform across the block
    for (int ci = 0; ci < nchunks; ++ci) {
        const int koff = ci * 32;

        // ---- cooperative staging of K (contiguous 8KB) and V^T rows ----
        {
            const unsigned short* gk = kbase + (size_t)koff * HD + tid * 32;
            unsigned short*       lk = &Kl[tid * 32];
            lds_copy16(gk,      lk);
            lds_copy16(gk + 8,  lk + 8);
            lds_copy16(gk + 16, lk + 16);
            lds_copy16(gk + 24, lk + 24);
            const unsigned short* gv = vbase + (size_t)tid * SS + koff;
            unsigned short*       lv = &Vl[tid * 32];
            lds_copy16(gv,      lv);
            lds_copy16(gv + 8,  lv + 8);
            lds_copy16(gv + 16, lv + 16);
            lds_copy16(gv + 24, lv + 24);
        }
        wait_lds_copies();
        __syncthreads();

        if (koff <= q0 + 15) {   // this wave still has unmasked keys in the chunk
            // ---- S tile = Q * K^T  (16 x 32, two 16x16 halves) ----
            v8f sv[2];
#pragma unroll
            for (int nh = 0; nh < 2; ++nh) {
                v8f acc = {};
#pragma unroll
                for (int t = 0; t < 4; ++t) {
                    Frag bk;
                    const unsigned short* p = &Kl[(nh * 16 + l15) * HD + t * 32 + lh * 8];
                    bk.q[0] = *(const v4u*)p;
                    bk.q[1] = *(const v4u*)(p + 16);
                    acc = __builtin_amdgcn_wmma_f32_16x16x32_bf16(
                        false, aq[t].bf, false, bk.bf, (short)0, acc, false, false);
                }
                sv[nh] = acc;
            }

            // ---- online softmax (C layout: row m = r + 8*lh, col n = l15) ----
            const int key0 = koff + l15;
            const int key1 = koff + 16 + l15;
#pragma unroll
            for (int r = 0; r < 8; ++r) {
                int m = r + (lh << 3);
                int q = q0 + m;
                float v0 = (key0 <= q) ? sv[0][r] * scale : -__builtin_inff();
                float v1 = (key1 <= q) ? sv[1][r] * scale : -__builtin_inff();
                float cm = fmaxf(v0, v1);
                cm = fmaxf(cm, __shfl_xor(cm, 1));
                cm = fmaxf(cm, __shfl_xor(cm, 2));
                cm = fmaxf(cm, __shfl_xor(cm, 4));
                cm = fmaxf(cm, __shfl_xor(cm, 8));
                float mn = fmaxf(mrow[r], cm);
                float a  = __expf(mrow[r] - mn);
                float p0 = __expf(v0 - mn);
                float p1 = __expf(v1 - mn);
                float cs = p0 + p1;
                cs += __shfl_xor(cs, 1);
                cs += __shfl_xor(cs, 2);
                cs += __shfl_xor(cs, 4);
                cs += __shfl_xor(cs, 8);
                l[r]     = l[r] * a + cs;
                mrow[r]  = mn;
                alpha[r] = a;
                pbuf[w][m * 32 + l15]      = f2b(p0);
                pbuf[w][m * 32 + 16 + l15] = f2b(p1);
            }

            // rescale running output by alpha (row mapping identical to C layout)
#pragma unroll
            for (int t = 0; t < 8; ++t)
#pragma unroll
                for (int r = 0; r < 8; ++r) o[t][r] *= alpha[r];

            // reload P in A-operand layout from the per-wave LDS tile
            Frag pa;
            {
                const unsigned short* pp = &pbuf[w][l15 * 32 + lh * 8];
                pa.q[0] = *(const v4u*)pp;
                pa.q[1] = *(const v4u*)(pp + 16);
            }

            // ---- O += P * V ----
#pragma unroll
            for (int t = 0; t < 8; ++t) {
                Frag bv;
                const unsigned short* p = &Vl[(t * 16 + l15) * 32 + lh * 8];
                bv.q[0] = *(const v4u*)p;
                bv.q[1] = *(const v4u*)(p + 16);
                o[t] = __builtin_amdgcn_wmma_f32_16x16x32_bf16(
                    false, pa.bf, false, bv.bf, (short)0, o[t], false, false);
            }
        }
        __syncthreads();   // protect K/V LDS tiles before next chunk overwrites
    }

    // ---- normalize and write bf16 output row-major [B*S][DIM] ----
#pragma unroll
    for (int t = 0; t < 8; ++t)
#pragma unroll
        for (int r = 0; r < 8; ++r) {
            int m = r + (lh << 3);
            int q = q0 + m;
            float v = o[t][r] / l[r];
            att[(size_t)(b * SS + q) * DIMD + h * HD + t * 16 + l15] = f2b(v);
        }
}

// ---------------- host side ----------------
static inline long cdiv(long a, long b) { return (a + b - 1) / b; }

extern "C" void kernel_launch(void* const* d_in, const int* in_sizes, int n_in,
                              void* d_out, int out_size, void* d_ws, size_t ws_size,
                              hipStream_t stream) {
    const float* x    = (const float*)d_in[0];
    const float* fcos = (const float*)d_in[1];
    const float* fsin = (const float*)d_in[2];
    const float* wq   = (const float*)d_in[3];
    const float* wk   = (const float*)d_in[4];
    const float* wv   = (const float*)d_in[5];
    const float* wo   = (const float*)d_in[6];
    float* out = (float*)d_out;

    // workspace layout (bf16 = unsigned short elements)
    unsigned short* p = (unsigned short*)d_ws;
    const size_t Mi = 1024 * 1024;
    unsigned short* xb    = p;                // 8 Mi elems  [M][K]
    unsigned short* wqkvT = xb    + 8 * Mi;   // 8 Mi        [NQKV][K] = wqT|wkT|wvT
    unsigned short* wqT   = wqkvT;            //   [NQ][K]
    unsigned short* wkT   = wqT   + 4 * Mi;   //   [NKV][K]
    unsigned short* wvT   = wkT   + 2 * Mi;   //   [NKV][K]
    unsigned short* woT   = wqkvT + 8 * Mi;   // 4 Mi        [DIM][K]
    unsigned short* qkv   = woT   + 4 * Mi;   // 16 Mi       [M][NQKV]
    unsigned short* Qa    = qkv   + 16 * Mi;  // 8 Mi        [B][H][S][HD]
    unsigned short* Ka    = Qa    + 8 * Mi;   // 4 Mi        [B][KV][S][HD]
    unsigned short* Vt    = Ka    + 4 * Mi;   // 4 Mi        [B][KV][HD][S]
    unsigned short* att   = Vt    + 4 * Mi;   // 8 Mi        [M][DIM]
    (void)ws_size; (void)in_sizes; (void)n_in; (void)out_size;

    const int T = 256;

    // 1) fp32 -> bf16 conversions (weights transposed; q|k|v packed contiguously)
    conv_kernel<<<cdiv((long)MTOT * DIMD, T), T, 0, stream>>>(x, xb, (long)MTOT * DIMD);
    convT_kernel<<<cdiv((long)DIMD * NQ,  T), T, 0, stream>>>(wq, wqT, DIMD, NQ);
    convT_kernel<<<cdiv((long)DIMD * NKV, T), T, 0, stream>>>(wk, wkT, DIMD, NKV);
    convT_kernel<<<cdiv((long)DIMD * NKV, T), T, 0, stream>>>(wv, wvT, DIMD, NKV);
    convT_kernel<<<cdiv((long)DIMD * DIMD,T), T, 0, stream>>>(wo, woT, DIMD, DIMD);

    // 2) fused QKV projection (single bf16 WMMA GEMM, bf16 out)
    gemm_bf16<1><<<dim3(NQKV / 128, MTOT / 128), T, 0, stream>>>(xb, wqkvT, qkv,
                                                                 MTOT, NQKV, DIMD);

    // 3) RoPE + layout rearrangement from the fused buffer
    rope_kernel<<<cdiv((long)BB * SS * HH  * HALF, T), T, 0, stream>>>(qkv, NQKV, 0,
                                                                       fcos, fsin, Qa, HH);
    rope_kernel<<<cdiv((long)BB * SS * KVH * HALF, T), T, 0, stream>>>(qkv, NQKV, NQ,
                                                                       fcos, fsin, Ka, KVH);
    vtrans_kernel<<<cdiv((long)BB * SS * KVH * HD, T), T, 0, stream>>>(qkv, NQKV, NQ + NKV, Vt);

    // 4) causal flash attention (WMMA QK^T and PV; K/V staged in LDS, async path)
    attn_kernel<<<dim3(SS / 64, BB * HH), 128, 0, stream>>>(Qa, Ka, Vt, att);

    // 5) output projection, fp32 result
    gemm_bf16<0><<<dim3(DIMD / 128, MTOT / 128), T, 0, stream>>>(att, woT, out,
                                                                 MTOT, DIMD, DIMD);
}
